// SelfAttention_49950469653031
// MI455X (gfx1250) — compile-verified, matrix operand order checked by reading, not run on hardware
//
#include <hip/hip_runtime.h>

// ---------------------------------------------------------------------------
// Self-attention for MI455X (gfx1250): f16 WMMA path, f32 accumulation.
// x[8,2048,1024] -> Q,K,V[.,.,64] -> causal+pad masked softmax -> out (fp32).
// All GEMMs go through v_wmma_f32_16x16x32_f16 (wave32).
// ---------------------------------------------------------------------------

typedef __attribute__((ext_vector_type(16))) _Float16 v16h;
typedef __attribute__((ext_vector_type(8)))  _Float16 v8h;
typedef __attribute__((ext_vector_type(8)))  float    v8f;
typedef __attribute__((ext_vector_type(4)))  float    v4f;

#define WMMA_F16(A, B, C) \
  __builtin_amdgcn_wmma_f32_16x16x32_f16(false, (A), false, (B), (short)0, (C), false, false)

constexpr int Bb = 8, S = 2048, Hh = 1024, DK = 64;

// ---------------------------------------------------------------------------
// Kernel 0: x fp32 -> f16, 8 elements/thread (2x b128 loads, 1x b128 store)
// ---------------------------------------------------------------------------
__global__ void cvt_x(const float* __restrict__ x, _Float16* __restrict__ xh, int n8) {
  int i = blockIdx.x * blockDim.x + threadIdx.x;
  if (i >= n8) return;
  const v4f* src = (const v4f*)x + (size_t)i * 2;
  v4f a = src[0], b = src[1];
  v8h o;
  o[0] = (_Float16)a[0]; o[1] = (_Float16)a[1];
  o[2] = (_Float16)a[2]; o[3] = (_Float16)a[3];
  o[4] = (_Float16)b[0]; o[5] = (_Float16)b[1];
  o[6] = (_Float16)b[2]; o[7] = (_Float16)b[3];
  ((v8h*)xh)[i] = o;
}

// ---------------------------------------------------------------------------
// Kernel 1: W [H,64] fp32 -> WT [64,H] f16 (so WMMA B-operand is K-contiguous)
// ---------------------------------------------------------------------------
__global__ void wtrans(const float* __restrict__ W, _Float16* __restrict__ WT) {
  int i = blockIdx.x * blockDim.x + threadIdx.x;  // over H*DK
  if (i >= Hh * DK) return;
  int h = i >> 6, d = i & 63;
  WT[(size_t)d * Hh + h] = (_Float16)W[i];
}

// ---------------------------------------------------------------------------
// Kernel 2: QKV projection. One block (4 waves) per 16-row tile of S.
// Wave w computes DK columns [16w,16w+16) of Q, K and V via WMMA over H.
// V is stored transposed (Vt[64][S]) so the P*V B-operand is key-contiguous.
// ---------------------------------------------------------------------------
__global__ __launch_bounds__(128) void qkv_proj(
    const _Float16* __restrict__ xh,
    const _Float16* __restrict__ WqT, const _Float16* __restrict__ WkT,
    const _Float16* __restrict__ WvT,
    const float* __restrict__ bq, const float* __restrict__ bk,
    const float* __restrict__ bv,
    _Float16* __restrict__ Qh, _Float16* __restrict__ Kh,
    _Float16* __restrict__ Vt) {
  const int tid = threadIdx.x, wave = tid >> 5, lane = tid & 31;
  const int b = blockIdx.x >> 7, st = blockIdx.x & 127;  // 128 row-tiles per batch
  const int s0 = st << 4, n0 = wave << 4;
  const int m = lane & 15, hg = lane >> 4;

  const _Float16* ap = xh + ((size_t)(b * S + s0 + m)) * Hh + 16 * hg;
  const _Float16* qp = WqT + (size_t)(n0 + m) * Hh + 16 * hg;
  const _Float16* kp = WkT + (size_t)(n0 + m) * Hh + 16 * hg;
  const _Float16* vp = WvT + (size_t)(n0 + m) * Hh + 16 * hg;

  v8f aq = {}, ak = {}, av = {};
#pragma unroll 2
  for (int kc = 0; kc < Hh / 32; kc++) {
    v16h A  = *(const v16h*)(ap + kc * 32);
    v16h Bq = *(const v16h*)(qp + kc * 32);
    v16h Bk = *(const v16h*)(kp + kc * 32);
    v16h Bv = *(const v16h*)(vp + kc * 32);
    aq = WMMA_F16(A, Bq, aq);
    ak = WMMA_F16(A, Bk, ak);
    av = WMMA_F16(A, Bv, av);
  }

  const float bqv = bq[n0 + m], bkv = bk[n0 + m], bvv = bv[n0 + m];

  // Q, K: row-major [S,64] f16 stores (C-layout: row = r + 8*hg, col = m)
  _Float16* qo = Qh + ((size_t)(b * S + s0)) * DK + n0 + m;
  _Float16* ko = Kh + ((size_t)(b * S + s0)) * DK + n0 + m;
#pragma unroll
  for (int r = 0; r < 8; r++) {
    int row = r + 8 * hg;
    qo[(size_t)row * DK] = (_Float16)(aq[r] + bqv);
    ko[(size_t)row * DK] = (_Float16)(ak[r] + bkv);
  }
  // V: transposed store; 8 consecutive rows per lane -> one 16-byte store
  v8h pv;
#pragma unroll
  for (int r = 0; r < 8; r++) pv[r] = (_Float16)(av[r] + bvv);
  *(v8h*)(Vt + ((size_t)(b * DK + n0 + m)) * S + s0 + 8 * hg) = pv;
}

// ---------------------------------------------------------------------------
// Kernel 3: flash-attention. One wave per 16-query tile; 32-key blocks.
// Online softmax with wave-internal shfl_xor row reductions; P re-layout
// through LDS (same-wave RAW guarded by s_wait_dscnt 0).
// ---------------------------------------------------------------------------
__global__ __launch_bounds__(128) void attn(
    const _Float16* __restrict__ Qh, const _Float16* __restrict__ Kh,
    const _Float16* __restrict__ Vt, const unsigned char* __restrict__ pad,
    float* __restrict__ out) {
  __shared__ _Float16 Pl[4][16 * 32];
  const int tid = threadIdx.x, wave = tid >> 5, lane = tid & 31;
  const int gt = blockIdx.x * 4 + wave;       // 1024 query tiles total
  const int b = gt >> 7, qt = gt & 127;
  const int q0 = qt << 4;
  const int m = lane & 15, hg = lane >> 4;
  _Float16* P = &Pl[wave][0];

  // Q A-operand, both K-dim chunks, loaded once (lane: row m, d contiguous)
  const _Float16* qrow = Qh + ((size_t)(b * S) + q0 + m) * DK + 16 * hg;
  const v16h A0 = *(const v16h*)qrow;
  const v16h A1 = *(const v16h*)(qrow + 32);

  v8f O0 = {}, O1 = {}, O2 = {}, O3 = {};
  float mr[8], lr[8];
#pragma unroll
  for (int r = 0; r < 8; r++) { mr[r] = -3.0e38f; lr[r] = 0.f; }

  const unsigned char* padb = pad + b * S;
  const _Float16* vb = Vt + (size_t)b * DK * S;
  const int nkb = (q0 + 15) / 32 + 1;         // wave-uniform causal bound

  for (int kb = 0; kb < nkb; kb++) {
    const int k0 = kb * 32;
    // scores: two 16-key subtiles, K-dim = d = 64 (2 WMMA chunks each)
    const _Float16* kr0 = Kh + ((size_t)(b * S) + k0 + m) * DK + 16 * hg;
    v8f S0 = {}, S1 = {};
    {
      v16h B0 = *(const v16h*)kr0;                 S0 = WMMA_F16(A0, B0, S0);
      v16h B1 = *(const v16h*)(kr0 + 32);          S0 = WMMA_F16(A1, B1, S0);
      v16h B2 = *(const v16h*)(kr0 + 16 * DK);     S1 = WMMA_F16(A0, B2, S1);
      v16h B3 = *(const v16h*)(kr0 + 16 * DK + 32);S1 = WMMA_F16(A1, B3, S1);
    }
    const int key0 = k0 + m, key1 = key0 + 16;
    const bool p0 = padb[key0] != 0, p1 = padb[key1] != 0;

#pragma unroll
    for (int r = 0; r < 8; r++) {
      const int q = q0 + r + 8 * hg;
      float s0 = (p0 || key0 > q) ? -3.0e38f : S0[r] * 0.125f;  // 1/sqrt(64)
      float s1 = (p1 || key1 > q) ? -3.0e38f : S1[r] * 0.125f;
      float bm = fmaxf(s0, s1);
      bm = fmaxf(bm, __shfl_xor(bm, 1));
      bm = fmaxf(bm, __shfl_xor(bm, 2));
      bm = fmaxf(bm, __shfl_xor(bm, 4));
      bm = fmaxf(bm, __shfl_xor(bm, 8));
      const float mn = fmaxf(mr[r], bm);
      const float f = __expf(mr[r] - mn);
      mr[r] = mn;
      const float e0 = __expf(s0 - mn), e1 = __expf(s1 - mn);
      float rs = e0 + e1;
      rs += __shfl_xor(rs, 1);
      rs += __shfl_xor(rs, 2);
      rs += __shfl_xor(rs, 4);
      rs += __shfl_xor(rs, 8);
      lr[r] = lr[r] * f + rs;
      O0[r] *= f; O1[r] *= f; O2[r] *= f; O3[r] *= f;
      const int row = r + 8 * hg;
      P[row * 32 + m]      = (_Float16)e0;   // C-layout -> [row][key] in LDS
      P[row * 32 + 16 + m] = (_Float16)e1;
    }
    // same-wave LDS RAW: wait for DS stores before re-reading as A-operand
    asm volatile("s_wait_dscnt 0" ::: "memory");
    const v16h PA = *(const v16h*)(P + m * 32 + 16 * hg);
    {
      v16h V0 = *(const v16h*)(vb + (size_t)(0 * 16 + m) * S + k0 + 16 * hg);
      O0 = WMMA_F16(PA, V0, O0);
      v16h V1 = *(const v16h*)(vb + (size_t)(1 * 16 + m) * S + k0 + 16 * hg);
      O1 = WMMA_F16(PA, V1, O1);
      v16h V2 = *(const v16h*)(vb + (size_t)(2 * 16 + m) * S + k0 + 16 * hg);
      O2 = WMMA_F16(PA, V2, O2);
      v16h V3 = *(const v16h*)(vb + (size_t)(3 * 16 + m) * S + k0 + 16 * hg);
      O3 = WMMA_F16(PA, V3, O3);
    }
  }

  // epilogue: divide by softmax denom, zero padded query rows, fp32 store
#pragma unroll
  for (int r = 0; r < 8; r++) {
    const int row = r + 8 * hg, q = q0 + row;
    const float sc = padb[q] ? 0.f : 1.f / lr[r];
    float* op = out + ((size_t)(b * S) + q) * DK + m;
    op[0]  = O0[r] * sc;
    op[16] = O1[r] * sc;
    op[32] = O2[r] * sc;
    op[48] = O3[r] * sc;
  }
}

// ---------------------------------------------------------------------------
extern "C" void kernel_launch(void* const* d_in, const int* in_sizes, int n_in,
                              void* d_out, int out_size, void* d_ws, size_t ws_size,
                              hipStream_t stream) {
  (void)in_sizes; (void)n_in; (void)out_size; (void)ws_size;
  const float* x  = (const float*)d_in[0];
  const unsigned char* pad = (const unsigned char*)d_in[1];  // bool mask [B,S]
  const float* Wq = (const float*)d_in[2];
  const float* bq = (const float*)d_in[3];
  const float* Wk = (const float*)d_in[4];
  const float* bk = (const float*)d_in[5];
  const float* Wv = (const float*)d_in[6];
  const float* bv = (const float*)d_in[7];
  float* out = (float*)d_out;

  // workspace layout (f16): xh | WqT | WkT | WvT | Qh | Kh | Vt  (~40 MB)
  _Float16* ws = (_Float16*)d_ws;
  size_t o = 0;
  _Float16* xh  = ws + o; o += (size_t)Bb * S * Hh;
  _Float16* WqT = ws + o; o += (size_t)DK * Hh;
  _Float16* WkT = ws + o; o += (size_t)DK * Hh;
  _Float16* WvT = ws + o; o += (size_t)DK * Hh;
  _Float16* Qh  = ws + o; o += (size_t)Bb * S * DK;
  _Float16* Kh  = ws + o; o += (size_t)Bb * S * DK;
  _Float16* Vt  = ws + o;

  const int n8 = Bb * S * Hh / 8;
  cvt_x<<<(n8 + 255) / 256, 256, 0, stream>>>(x, xh, n8);
  const int nw = (Hh * DK + 255) / 256;
  wtrans<<<nw, 256, 0, stream>>>(Wq, WqT);
  wtrans<<<nw, 256, 0, stream>>>(Wk, WkT);
  wtrans<<<nw, 256, 0, stream>>>(Wv, WvT);
  qkv_proj<<<Bb * (S / 16), 128, 0, stream>>>(xh, WqT, WkT, WvT, bq, bk, bv,
                                              Qh, Kh, Vt);
  attn<<<Bb * (S / 16) / 4, 128, 0, stream>>>(Qh, Kh, Vt, pad, out);
}
